// StandardMultiHeadAttention_40827959116479
// MI455X (gfx1250) — compile-verified, matrix operand order checked by reading, not run on hardware
//
#include <hip/hip_runtime.h>

// ---------------------------------------------------------------------------
// StandardMultiHeadAttention forward for MI455X (gfx1250, wave32, WMMA).
// B=2, S=2048, D=1024, H=16, hd=64.  Outputs: out [B,S,D] fp32  ++
// attn_weights [B,H,S,S] fp32 (concatenated flat in d_out).
// bf16 WMMA (16x16x32) with fp32 accumulation; the 512MB attention tensor
// lives directly in d_out, softmax runs in-place on it, and the AV GEMM
// converts fp32->bf16 while staging to LDS.  bf16 GEMMs use async
// global->LDS loads (ASYNCcnt) with double-buffered LDS tiles.
// ---------------------------------------------------------------------------

typedef __attribute__((ext_vector_type(16))) __bf16 v16bf;
typedef __attribute__((ext_vector_type(8)))  float  v8f;
typedef __attribute__((ext_vector_type(4)))  int    v4i;

static constexpr int  S_LEN = 2048;
static constexpr int  DM    = 1024;
static constexpr int  NH    = 16;
static constexpr int  HD    = 64;
static constexpr int  BATCH = 2;
static constexpr long long QSZ = (long long)BATCH * NH * S_LEN * HD; // 4,194,304 elems

#if __has_builtin(__builtin_amdgcn_global_load_async_to_lds_b128)
#define MHA_HAVE_ASYNC_LDS 1
// builtin signature: (int4 AS1* src, int4 AS3* dst, imm offset, imm cpol)
typedef __attribute__((address_space(1))) v4i* gptr_b128;
typedef __attribute__((address_space(3))) v4i* lptr_b128;
__device__ __forceinline__ void async_b128(const void* g, void* l) {
    __builtin_amdgcn_global_load_async_to_lds_b128((gptr_b128)g, (lptr_b128)l, 0, 0);
}
#else
#define MHA_HAVE_ASYNC_LDS 0
#endif

template <int N>
__device__ __forceinline__ void wait_asynccnt() {
#if __has_builtin(__builtin_amdgcn_s_wait_asynccnt)
    __builtin_amdgcn_s_wait_asynccnt((short)N);
#else
    asm volatile("s_wait_asynccnt %0" ::"i"(N) : "memory");
#endif
}

__device__ __forceinline__ unsigned short f2bf(float f) {
    unsigned u = __builtin_bit_cast(unsigned, f);
    u += 0x7FFFu + ((u >> 16) & 1u);          // round-to-nearest-even
    return (unsigned short)(u >> 16);
}
__device__ __forceinline__ unsigned pack2(float lo, float hi) {
    return (unsigned)f2bf(lo) | ((unsigned)f2bf(hi) << 16);
}

union FragBF { uint4 u4[2]; v16bf v; };

// ---------------------------------------------------------------------------
// fp32 -> bf16 elementwise convert (8 elems / thread, all sizes % 2048 == 0)
// ---------------------------------------------------------------------------
__global__ __launch_bounds__(256) void f32_to_bf16_kernel(const float* __restrict__ in,
                                                          unsigned short* __restrict__ out,
                                                          int n) {
    int idx = (blockIdx.x * 256 + threadIdx.x) * 8;
    if (idx >= n) return;
    float4 a = *(const float4*)(in + idx);
    float4 b = *(const float4*)(in + idx + 4);
    uint4 o;
    o.x = pack2(a.x, a.y); o.y = pack2(a.z, a.w);
    o.z = pack2(b.x, b.y); o.w = pack2(b.z, b.w);
    *(uint4*)(out + idx) = o;
}

// ---------------------------------------------------------------------------
// Generic tiled bf16 WMMA GEMM:  C[m,n] = alpha * sum_k A[m,k] * B[n,k]
// (A @ B^T, both row-major).  BM=BN=128, BK=32, 256 threads = 8 wave32s,
// each wave computes a 64x32 patch = 4x2 grid of 16x16x32 WMMA tiles.
// Epilogue by MODE:
//   MODE 1 (QKV): N=3072 fused QKV; scatter Q,K per-head row-major and V
//                 transposed [b,h,hd,S] into workspace (bf16).
//   MODE 2 (F32): fp32 store, C + z*sCb, ld = N, scaled by alpha.
//   MODE 3 (AV) : bf16 store into head-concat layout [B*S, DM], guard col<N.
// AF32: A operand is fp32 (attention probs) -> bf16 while staging into LDS.
// bf16 path (AF32=false): requires M%128==0 && N%128==0 (true for all
// launches); uses async global->LDS double buffering when available.
// ---------------------------------------------------------------------------
template <int MODE, bool AF32>
__global__ __launch_bounds__(256) void gemm_wmma_bf16(const void* __restrict__ Av,
                                                      const unsigned short* __restrict__ Bm,
                                                      void* __restrict__ Cv,
                                                      int M, int N, int K,
                                                      long long sAb, long long sBb, long long sCb,
                                                      float alpha) {
    constexpr int BM = 128, BN = 128, BK = 32;
    constexpr int LDT = 40;                        // padded row (80B, 16B aligned)
    __shared__ alignas(16) unsigned short sA[2][BM * LDT];
    __shared__ alignas(16) unsigned short sB[2][BN * LDT];

    const int tid   = threadIdx.x;
    const int lane  = tid & 31;
    const int wid   = tid >> 5;
    const int waveM = wid & 1;                     // 2 row-groups of 64
    const int waveN = wid >> 1;                    // 4 col-groups of 32
    const int z     = blockIdx.z;
    const int mBase = blockIdx.y * BM;
    const int nBase = blockIdx.x * BN;

    const unsigned short* Abf = nullptr;
    const float*          Af  = nullptr;
    if constexpr (AF32) Af  = (const float*)Av          + (size_t)z * sAb;
    else                Abf = (const unsigned short*)Av + (size_t)z * sAb;
    const unsigned short* Bp = Bm + (size_t)z * sBb;

    v8f acc[4][2];
    #pragma unroll
    for (int i = 0; i < 4; ++i)
        #pragma unroll
        for (int j = 0; j < 2; ++j)
            acc[i][j] = {};

    const int kTiles = K / BK;

    // per-thread staging coordinates (16B chunks of a 128x32 bf16 tile)
    const int cA0  = tid;                 // chunk ids: tid, tid+256
    const int rowA0 = cA0 >> 2,  kcA0 = (cA0 & 3) * 8;
    const int rowA1 = (cA0 + 256) >> 2, kcA1 = ((cA0 + 256) & 3) * 8;

    // fragment-load coordinates (ISA 7.12.2 16-bit layouts)
    const int lrow = lane & 15;
    const int koff = (lane >> 4) * 8;

    auto mma_tile = [&](const unsigned short* bufA, const unsigned short* bufB) {
        FragBF a[4], b[2];
        #pragma unroll
        for (int i = 0; i < 4; ++i) {
            const unsigned short* p = &bufA[(waveM * 64 + i * 16 + lrow) * LDT + koff];
            a[i].u4[0] = *(const uint4*)p;          // ds_load_b128
            a[i].u4[1] = *(const uint4*)(p + 16);   // ds_load_b128
        }
        #pragma unroll
        for (int j = 0; j < 2; ++j) {
            const unsigned short* p = &bufB[(waveN * 32 + j * 16 + lrow) * LDT + koff];
            b[j].u4[0] = *(const uint4*)p;
            b[j].u4[1] = *(const uint4*)(p + 16);
        }
        #pragma unroll
        for (int i = 0; i < 4; ++i)
            #pragma unroll
            for (int j = 0; j < 2; ++j)
                acc[i][j] = __builtin_amdgcn_wmma_f32_16x16x32_bf16(
                    false, a[i].v, false, b[j].v, (short)0, acc[i][j], false, false);
    };

    if constexpr (!AF32) {
#if MHA_HAVE_ASYNC_LDS
        // ---- async global->LDS, double-buffered: 4 async ops / thread / tile
        auto issue_tile = [&](int kb, int buf) {
            const size_t ka = (size_t)kb * BK;
            async_b128(Abf + (size_t)(mBase + rowA0) * K + ka + kcA0,
                       &sA[buf][rowA0 * LDT + kcA0]);
            async_b128(Abf + (size_t)(mBase + rowA1) * K + ka + kcA1,
                       &sA[buf][rowA1 * LDT + kcA1]);
            async_b128(Bp + (size_t)(nBase + rowA0) * K + ka + kcA0,
                       &sB[buf][rowA0 * LDT + kcA0]);
            async_b128(Bp + (size_t)(nBase + rowA1) * K + ka + kcA1,
                       &sB[buf][rowA1 * LDT + kcA1]);
        };
        issue_tile(0, 0);
        for (int kb = 0; kb < kTiles; ++kb) {
            const int cur = kb & 1;
            if (kb + 1 < kTiles) {
                issue_tile(kb + 1, cur ^ 1);
                wait_asynccnt<4>();      // async loads retire in order: tile kb done
            } else {
                wait_asynccnt<0>();
            }
            __syncthreads();             // all waves' tile-kb writes visible
            mma_tile(sA[cur], sB[cur]);
            __syncthreads();             // reads done before buffer reuse
        }
#else
        // ---- fallback: register staging, single buffer
        for (int kb = 0; kb < kTiles; ++kb) {
            #pragma unroll
            for (int it = 0; it < 2; ++it) {
                int c = tid + it * 256, row = c >> 2, kc = (c & 3) * 8;
                const uint4 va = *(const uint4*)(Abf + (size_t)(mBase + row) * K + kb * BK + kc);
                *(uint4*)&sA[0][row * LDT + kc] = va;
                const uint4 vb = *(const uint4*)(Bp + (size_t)(nBase + row) * K + kb * BK + kc);
                *(uint4*)&sB[0][row * LDT + kc] = vb;
                if (kb + 1 < kTiles) {
                    __builtin_prefetch(Abf + (size_t)(mBase + row) * K + (kb + 1) * BK + kc, 0, 0);
                    __builtin_prefetch(Bp + (size_t)(nBase + row) * K + (kb + 1) * BK + kc, 0, 0);
                }
            }
            __syncthreads();
            mma_tile(sA[0], sB[0]);
            __syncthreads();
        }
#endif
    } else {
        // ---- fp32 A operand (attention probs): convert to bf16 in staging;
        //      B rows guarded (AV launch has N=64 < BN).
        for (int kb = 0; kb < kTiles; ++kb) {
            #pragma unroll
            for (int it = 0; it < 4; ++it) {
                int c = tid + it * 256, row = c >> 3, kc = (c & 7) * 4;
                const float4 f = *(const float4*)(Af + (size_t)(mBase + row) * K + kb * BK + kc);
                uint2 p;
                p.x = pack2(f.x, f.y);
                p.y = pack2(f.z, f.w);
                *(uint2*)&sA[0][row * LDT + kc] = p;
            }
            #pragma unroll
            for (int it = 0; it < 2; ++it) {
                int c = tid + it * 256, row = c >> 2, kc = (c & 3) * 8;
                uint4 v = make_uint4(0u, 0u, 0u, 0u);
                if (nBase + row < N)
                    v = *(const uint4*)(Bp + (size_t)(nBase + row) * K + kb * BK + kc);
                *(uint4*)&sB[0][row * LDT + kc] = v;
            }
            __syncthreads();
            mma_tile(sA[0], sB[0]);
            __syncthreads();
        }
    }

    // ---- epilogue: C/D layout (7.12.2): vgpr r -> M = r + 8*(lane>=16), N = lane&15
    const int lm = (lane >> 4) * 8;
    const int ln = lane & 15;
    #pragma unroll
    for (int i = 0; i < 4; ++i) {
        #pragma unroll
        for (int j = 0; j < 2; ++j) {
            const int rowb = mBase + waveM * 64 + i * 16 + lm;
            const int col  = nBase + waveN * 32 + j * 16 + ln;
            #pragma unroll
            for (int r = 0; r < 8; ++r) {
                const float val = alpha * acc[i][j][r];
                const int   row = rowb + r;
                if constexpr (MODE == 2) {                       // fp32 row-major (+batch)
                    ((float*)Cv)[(size_t)z * sCb + (size_t)row * N + col] = val;
                } else if constexpr (MODE == 1) {                // fused QKV scatter
                    unsigned short* qp = (unsigned short*)Cv;
                    const int mat = col >> 10;
                    const int h   = (col >> 6) & 15;
                    const int d   = col & 63;
                    const int bb  = row >> 11;
                    const int ss  = row & 2047;
                    const unsigned short bv = f2bf(val);
                    if (mat == 0) {
                        qp[((size_t)((bb * NH + h) * S_LEN + ss)) * HD + d] = bv;
                    } else if (mat == 1) {
                        qp[QSZ + ((size_t)((bb * NH + h) * S_LEN + ss)) * HD + d] = bv;
                    } else {                                     // V stored transposed [b,h,hd,S]
                        qp[2 * QSZ + ((size_t)((bb * NH + h) * HD + d)) * S_LEN + ss] = bv;
                    }
                } else {                                         // MODE 3: AV -> head concat
                    if (col < N) {
                        const int bb = z >> 4;
                        const int h  = z & 15;
                        ((unsigned short*)Cv)[((size_t)(bb * S_LEN + row)) * DM + h * HD + col] =
                            f2bf(val);
                    }
                }
            }
        }
    }
}

// ---------------------------------------------------------------------------
// Row softmax, in place on fp32 scores (row length 2048), numerically stable.
// ---------------------------------------------------------------------------
__global__ __launch_bounds__(256) void softmax_rows_kernel(float* __restrict__ attn) {
    const size_t row = blockIdx.x;
    float* p = attn + row * (size_t)S_LEN;
    const int tid = threadIdx.x;
    __shared__ float red[256];

    float4 v0 = *(const float4*)(p + tid * 8);
    float4 v1 = *(const float4*)(p + tid * 8 + 4);

    float lmax = fmaxf(fmaxf(fmaxf(v0.x, v0.y), fmaxf(v0.z, v0.w)),
                       fmaxf(fmaxf(v1.x, v1.y), fmaxf(v1.z, v1.w)));
    red[tid] = lmax;
    __syncthreads();
    for (int s = 128; s > 0; s >>= 1) {
        if (tid < s) red[tid] = fmaxf(red[tid], red[tid + s]);
        __syncthreads();
    }
    const float m = red[0];
    __syncthreads();

    v0.x = __expf(v0.x - m); v0.y = __expf(v0.y - m);
    v0.z = __expf(v0.z - m); v0.w = __expf(v0.w - m);
    v1.x = __expf(v1.x - m); v1.y = __expf(v1.y - m);
    v1.z = __expf(v1.z - m); v1.w = __expf(v1.w - m);
    float lsum = (v0.x + v0.y + v0.z + v0.w) + (v1.x + v1.y + v1.z + v1.w);
    red[tid] = lsum;
    __syncthreads();
    for (int s = 128; s > 0; s >>= 1) {
        if (tid < s) red[tid] += red[tid + s];
        __syncthreads();
    }
    const float inv = 1.0f / red[0];

    v0.x *= inv; v0.y *= inv; v0.z *= inv; v0.w *= inv;
    v1.x *= inv; v1.y *= inv; v1.z *= inv; v1.w *= inv;
    *(float4*)(p + tid * 8)     = v0;
    *(float4*)(p + tid * 8 + 4) = v1;
}

// ---------------------------------------------------------------------------
// Orchestration. Workspace layout (bf16 elems):
//   xbf[4096*1024] | wqkv[3072*1024] | wo[1024*1024] | q[QSZ] | k[QSZ] |
//   vt[QSZ] | out_concat[4096*1024]     (~48 MB total)
// ---------------------------------------------------------------------------
extern "C" void kernel_launch(void* const* d_in, const int* in_sizes, int n_in,
                              void* d_out, int out_size, void* d_ws, size_t ws_size,
                              hipStream_t stream) {
    (void)in_sizes; (void)n_in; (void)out_size; (void)ws_size;

    const float* x  = (const float*)d_in[0];
    const float* Wq = (const float*)d_in[1];
    const float* Wk = (const float*)d_in[2];
    const float* Wv = (const float*)d_in[3];
    const float* Wo = (const float*)d_in[4];

    float* out  = (float*)d_out;                         // [4096, 1024]
    float* attn = out + (size_t)4096 * 1024;             // [2,16,2048,2048]

    unsigned short* xbf  = (unsigned short*)d_ws;
    unsigned short* wqkv = xbf  + (size_t)4096 * 1024;
    unsigned short* wo   = wqkv + (size_t)3072 * 1024;
    unsigned short* q    = wo   + (size_t)1024 * 1024;
    unsigned short* k    = q    + QSZ;
    unsigned short* vt   = k    + QSZ;
    unsigned short* oc   = vt   + QSZ;                   // head-concat attn output

    // 1) fp32 -> bf16 conversions (Wq|Wk|Wv packed into one 3072x1024 weight)
    f32_to_bf16_kernel<<<(4096 * 1024) / 2048, 256, 0, stream>>>(x,  xbf, 4096 * 1024);
    f32_to_bf16_kernel<<<(1024 * 1024) / 2048, 256, 0, stream>>>(Wq, wqkv,                   1024 * 1024);
    f32_to_bf16_kernel<<<(1024 * 1024) / 2048, 256, 0, stream>>>(Wk, wqkv + 1024 * 1024,     1024 * 1024);
    f32_to_bf16_kernel<<<(1024 * 1024) / 2048, 256, 0, stream>>>(Wv, wqkv + 2 * 1024 * 1024, 1024 * 1024);
    f32_to_bf16_kernel<<<(1024 * 1024) / 2048, 256, 0, stream>>>(Wo, wo, 1024 * 1024);

    // 2) fused QKV projection: [4096,1024] @ [3072,1024]^T -> per-head Q,K + V^T
    gemm_wmma_bf16<1, false><<<dim3(3072 / 128, 4096 / 128, 1), 256, 0, stream>>>(
        xbf, wqkv, q, 4096, 3072, 1024, 0, 0, 0, 1.0f);

    // 3) scores = (Q @ K^T) / sqrt(64), batched over 32 (b,h), fp32 into d_out
    gemm_wmma_bf16<2, false><<<dim3(2048 / 128, 2048 / 128, BATCH * NH), 256, 0, stream>>>(
        q, k, attn, 2048, 2048, 64,
        (long long)S_LEN * HD, (long long)S_LEN * HD, (long long)S_LEN * S_LEN, 0.125f);

    // 4) softmax rows in place (B*H*S = 65536 rows)
    softmax_rows_kernel<<<BATCH * NH * S_LEN, 256, 0, stream>>>(attn);

    // 5) out_h = attn @ V  (A = fp32 attn converted on the fly; B = V^T rows)
    gemm_wmma_bf16<3, true><<<dim3(1, 2048 / 128, BATCH * NH), 256, 0, stream>>>(
        attn, vt, oc, 2048, 64, 2048,
        (long long)S_LEN * S_LEN, (long long)HD * S_LEN, 0, 1.0f);

    // 6) final projection: [4096,1024] @ Wo^T -> d_out fp32
    gemm_wmma_bf16<2, false><<<dim3(1024 / 128, 4096 / 128, 1), 256, 0, stream>>>(
        oc, wo, out, 4096, 1024, 1024, 0, 0, 0, 1.0f);
}
